// LoRALayerWrapper_63677185130983
// MI455X (gfx1250) — matrix operand order repaired and checked by reading, not compiled, not run on hardware
//
#include <hip/hip_runtime.h>

// ---------------------------------------------------------------------------
// LoRA linear: out = x @ W^T + b + (x @ A) @ B^T
//   x: (8192, 4096) f32   W: (4096, 4096) f32   b: (4096,)
//   A: (4096, 16)   B: (4096, 16)
// Strategy: bf16 WMMA (v_wmma_f32_16x16x32_bf16) tiled GEMM, rank-16 epilogue.
// f32->bf16 conversion done with vector convert -> v_cvt_pk_bf16_f32 pairs.
// ---------------------------------------------------------------------------

typedef __attribute__((ext_vector_type(16))) __bf16 v16bf;
typedef __attribute__((ext_vector_type(16))) float  v16f;
typedef __attribute__((ext_vector_type(8)))  float  v8f;

#define DDIM   4096
#define MDIM   8192
#define RANK   16
#define TM     128
#define TN     128
#define TK     32
#define KSTEPS (DDIM / TK)          // 128
#define LDSS   80                   // bytes per LDS tile row (40 bf16, padded)

// -------------------- helpers --------------------

// 16 f32 -> 16 bf16 (32 bytes) into LDS; lowers to 8x v_cvt_pk_bf16_f32
__device__ inline void cvt_store16(void* dst, const float4* f) {
    union { float4 q[4]; v16f v; } in;
    in.q[0] = f[0]; in.q[1] = f[1]; in.q[2] = f[2]; in.q[3] = f[3];
    union { v16bf v; uint4 q[2]; } cv;
    cv.v = __builtin_convertvector(in.v, v16bf);
    ((uint4*)dst)[0] = cv.q[0];
    ((uint4*)dst)[1] = cv.q[1];
}

// A fragment (16x32 bf16, MxK). Lane l(0-15): row M=l, K = 0..15.
// Lane l(16-31): row M=l-16, K = 16..31.  base points at row 0 of tile.
__device__ inline v16bf load_frag_a(const unsigned char* base, int lane) {
    const int row = lane & 15;
    const int kb  = (lane >> 4) << 4;           // 0 or 16
    const unsigned char* p = base + row * LDSS;
    v16bf r;
    ((uint4*)&r)[0] = *(const uint4*)(p + kb * 2);
    ((uint4*)&r)[1] = *(const uint4*)(p + (kb + 8) * 2);
    return r;
}

// B fragment (32x16 bf16, KxN), B = W^T so B[k][n] = Wtile[n][k] (contiguous k).
// Lane l(0-15): col N=l, K=0..15. Lane l(16-31): col N=l-16, K=16..31.
__device__ inline v16bf load_frag_b(const unsigned char* base, int lane) {
    const int col = lane & 15;
    const int kb  = (lane >> 4) << 4;           // 0 or 16
    const unsigned char* p = base + col * LDSS + kb * 2;
    v16bf r;
    ((uint4*)&r)[0] = ((const uint4*)p)[0];
    ((uint4*)&r)[1] = ((const uint4*)p)[1];
    return r;
}

// -------------------- kernel 1: T = x @ A  (8192 x 16) --------------------

__global__ __launch_bounds__(128) void lora_xa_kernel(
    const float* __restrict__ x, const float* __restrict__ A,
    float* __restrict__ T)
{
    __shared__ float s[RANK];
    const int m = blockIdx.x;
    const int t = threadIdx.x;
    if (t < RANK) s[t] = 0.0f;
    __syncthreads();

    float acc[RANK];
#pragma unroll
    for (int r = 0; r < RANK; ++r) acc[r] = 0.0f;

    const float* xr = x + (size_t)m * DDIM;
    for (int k = t; k < DDIM; k += 128) {
        const float xv = xr[k];
        const float4* ar = (const float4*)(A + (size_t)k * RANK);
        const float4 a0 = ar[0], a1 = ar[1], a2 = ar[2], a3 = ar[3];
        acc[0]  += xv * a0.x; acc[1]  += xv * a0.y; acc[2]  += xv * a0.z; acc[3]  += xv * a0.w;
        acc[4]  += xv * a1.x; acc[5]  += xv * a1.y; acc[6]  += xv * a1.z; acc[7]  += xv * a1.w;
        acc[8]  += xv * a2.x; acc[9]  += xv * a2.y; acc[10] += xv * a2.z; acc[11] += xv * a2.w;
        acc[12] += xv * a3.x; acc[13] += xv * a3.y; acc[14] += xv * a3.z; acc[15] += xv * a3.w;
    }

    // wave32 tree reduce, then one LDS atomic per wave
#pragma unroll
    for (int r = 0; r < RANK; ++r) {
        float v = acc[r];
#pragma unroll
        for (int off = 16; off > 0; off >>= 1) v += __shfl_down(v, off, 32);
        if ((t & 31) == 0) atomicAdd(&s[r], v);
    }
    __syncthreads();
    if (t < RANK) T[(size_t)m * RANK + t] = s[t];
}

// -------------------- kernel 2: tiled bf16-WMMA GEMM + LoRA epilogue -------

__global__ __launch_bounds__(256) void lora_gemm_kernel(
    const float* __restrict__ x,   const float* __restrict__ Wm,
    const float* __restrict__ bvec, const float* __restrict__ loraB,
    const float* __restrict__ Tmat, float* __restrict__ out)
{
    __shared__ __align__(16) unsigned char smA[2][TM * LDSS];
    __shared__ __align__(16) unsigned char smB[2][TN * LDSS];

    const int tid  = threadIdx.x;
    const int lane = tid & 31;
    const int wave = tid >> 5;
    const int wm   = (wave & 3) * 32;   // wave M origin in tile
    const int wn   = (wave >> 2) * 64;  // wave N origin in tile

    const int n0 = blockIdx.x * TN;
    const int m0 = blockIdx.y * TM;

    // staging: 2 threads per row, 16 f32 each
    const int srow = tid >> 1;          // 0..127
    const int sseg = tid & 1;           // 0..1
    const float* gA = x  + (size_t)(m0 + srow) * DDIM + sseg * 16;
    const float* gB = Wm + (size_t)(n0 + srow) * DDIM + sseg * 16;
    const int soff = srow * LDSS + sseg * 32;

    v8f acc[2][4];
    const v8f vzero = {0.f, 0.f, 0.f, 0.f, 0.f, 0.f, 0.f, 0.f};
#pragma unroll
    for (int fm = 0; fm < 2; ++fm)
#pragma unroll
        for (int fn = 0; fn < 4; ++fn) acc[fm][fn] = vzero;

    float4 rA[4], rB[4];
    // stage tile 0 directly
    {
        const float4* pa = (const float4*)gA;
        const float4* pb = (const float4*)gB;
        rA[0] = pa[0]; rA[1] = pa[1]; rA[2] = pa[2]; rA[3] = pa[3];
        rB[0] = pb[0]; rB[1] = pb[1]; rB[2] = pb[2]; rB[3] = pb[3];
    }
    cvt_store16(&smA[0][soff], rA);
    cvt_store16(&smB[0][soff], rB);
    __syncthreads();
    // preload tile 1 into registers
    {
        const float4* pa = (const float4*)(gA + TK);
        const float4* pb = (const float4*)(gB + TK);
        rA[0] = pa[0]; rA[1] = pa[1]; rA[2] = pa[2]; rA[3] = pa[3];
        rB[0] = pb[0]; rB[1] = pb[1]; rB[2] = pb[2]; rB[3] = pb[3];
    }

    for (int ks = 0; ks < KSTEPS; ++ks) {
        const int cur = ks & 1;
        const int nxt = cur ^ 1;

        if (ks + 1 < KSTEPS) {          // commit prefetched tile ks+1
            cvt_store16(&smA[nxt][soff], rA);
            cvt_store16(&smB[nxt][soff], rB);
        }
        if (ks + 2 < KSTEPS) {          // issue global loads for tile ks+2
            const float4* pa = (const float4*)(gA + (ks + 2) * TK);
            const float4* pb = (const float4*)(gB + (ks + 2) * TK);
            rA[0] = pa[0]; rA[1] = pa[1]; rA[2] = pa[2]; rA[3] = pa[3];
            rB[0] = pb[0]; rB[1] = pb[1]; rB[2] = pb[2]; rB[3] = pb[3];
        }

        v16bf af[2], bfr[4];
#pragma unroll
        for (int fm = 0; fm < 2; ++fm)
            af[fm] = load_frag_a(&smA[cur][(wm + fm * 16) * LDSS], lane);
#pragma unroll
        for (int fn = 0; fn < 4; ++fn)
            bfr[fn] = load_frag_b(&smB[cur][(wn + fn * 16) * LDSS], lane);

#pragma unroll
        for (int fm = 0; fm < 2; ++fm)
#pragma unroll
            for (int fn = 0; fn < 4; ++fn)
                acc[fm][fn] = __builtin_amdgcn_wmma_f32_16x16x32_bf16(
                    false, af[fm], false, bfr[fn], (short)0, acc[fm][fn],
                    false, false);

        __syncthreads();
    }

    // ---- LoRA epilogue: acc += (T_tile 128x16) @ (loraB_tile 128x16)^T ----
    // stage as bf16 with K padded 16 -> 32 (upper half zero), reuse buffer 0
    {
        unsigned char* dA = &smA[0][srow * LDSS];
        unsigned char* dB = &smB[0][srow * LDSS];
        if (sseg == 0) {
            float4 tA[4], tB[4];
            const float4* pT = (const float4*)(Tmat  + (size_t)(m0 + srow) * RANK);
            const float4* pL = (const float4*)(loraB + (size_t)(n0 + srow) * RANK);
            tA[0] = pT[0]; tA[1] = pT[1]; tA[2] = pT[2]; tA[3] = pT[3];
            tB[0] = pL[0]; tB[1] = pL[1]; tB[2] = pL[2]; tB[3] = pL[3];
            cvt_store16(dA, tA);
            cvt_store16(dB, tB);
        } else {
            const uint4 z = {0u, 0u, 0u, 0u};
            ((uint4*)(dA + 32))[0] = z; ((uint4*)(dA + 48))[0] = z;
            ((uint4*)(dB + 32))[0] = z; ((uint4*)(dB + 48))[0] = z;
        }
    }
    __syncthreads();
    {
        v16bf af[2], bfr[4];
#pragma unroll
        for (int fm = 0; fm < 2; ++fm)
            af[fm] = load_frag_a(&smA[0][(wm + fm * 16) * LDSS], lane);
#pragma unroll
        for (int fn = 0; fn < 4; ++fn)
            bfr[fn] = load_frag_b(&smB[0][(wn + fn * 16) * LDSS], lane);
#pragma unroll
        for (int fm = 0; fm < 2; ++fm)
#pragma unroll
            for (int fn = 0; fn < 4; ++fn)
                acc[fm][fn] = __builtin_amdgcn_wmma_f32_16x16x32_bf16(
                    false, af[fm], false, bfr[fn], (short)0, acc[fm][fn],
                    false, false);
    }

    // ---- store: C/D layout is VGPR i, lane -> m = i + 8*(lane>=16), n = lane&15
    const int orow = (lane >> 4) * 8;
    const int ocol = lane & 15;
    float bias[4];
#pragma unroll
    for (int fn = 0; fn < 4; ++fn)
        bias[fn] = bvec[n0 + wn + fn * 16 + ocol];

#pragma unroll
    for (int fm = 0; fm < 2; ++fm)
#pragma unroll
        for (int fn = 0; fn < 4; ++fn)
#pragma unroll
            for (int i = 0; i < 8; ++i) {
                const size_t idx =
                    (size_t)(m0 + wm + fm * 16 + orow + i) * DDIM +
                    (size_t)(n0 + wn + fn * 16 + ocol);
                out[idx] = acc[fm][fn][i] + bias[fn];
            }
}

// -------------------- launcher --------------------

extern "C" void kernel_launch(void* const* d_in, const int* in_sizes, int n_in,
                              void* d_out, int out_size, void* d_ws, size_t ws_size,
                              hipStream_t stream) {
    const float* x     = (const float*)d_in[0];
    const float* W     = (const float*)d_in[1];
    const float* b     = (const float*)d_in[2];
    const float* loraA = (const float*)d_in[3];
    const float* loraB = (const float*)d_in[4];
    float* out = (float*)d_out;
    float* T   = (float*)d_ws;           // 8192 * 16 f32 = 512 KB scratch

    lora_xa_kernel<<<dim3(MDIM), dim3(128), 0, stream>>>(x, loraA, T);
    lora_gemm_kernel<<<dim3(DDIM / TN, MDIM / TM), dim3(256), 0, stream>>>(
        x, W, b, loraB, T, out);
}
